// NeuroSAT_14156212208242
// MI455X (gfx1250) — compile-verified, matrix operand order checked by reading, not run on hardware
//
#include <hip/hip_runtime.h>

// ---------------------------------------------------------------------------
// NeuroSAT forward for MI455X (gfx1250, wave32).
// Matmuls: V_WMMA_F32_16X16X32_BF16, f32 accumulate. Each wave computes a
// 16x128 output strip (8 WMMA tiles): one A fetch feeds 8 WMMAs; the shared
// B panel is staged per block into LDS (double buffered) via
// GLOBAL_LOAD_ASYNC_TO_LDS_B128 + s_wait_asynccnt. B tiles for a K-step are
// preloaded into distinct registers so ds_load latency pipelines with WMMA.
// x is {0,1} so bf16(x) is exact; bf16 x (128MB) stays L2-resident across the
// 16 big adjacency GEMMs. All LayerNorm / LSTM math in f32.
// ---------------------------------------------------------------------------

typedef __bf16 bf16_t;
typedef __attribute__((ext_vector_type(16))) __bf16 v16bf;
typedef __attribute__((ext_vector_type(8)))  __bf16 v8bf;
typedef __attribute__((ext_vector_type(8)))  float  v8f;
typedef __attribute__((ext_vector_type(4)))  int    v4i;

#define N_LITS    4096
#define N_VARS    2048
#define N_CLAUSES 16384
#define DDIM      128
#define N_ROUNDS  8

#define NT    8   // n-tiles per block strip (128 columns)
#define CHUNK 2   // k-tiles staged per LDS buffer (64 K-values, 16KB)

// ---------------- gfx1250 async global->LDS path ---------------------------
#if defined(__has_builtin)
#if __has_builtin(__builtin_amdgcn_global_load_async_to_lds_b128)
#define HAVE_ASYNC_LDS 1
#endif
#endif
#ifndef HAVE_ASYNC_LDS
#define HAVE_ASYNC_LDS 0
#endif

#if HAVE_ASYNC_LDS
typedef __attribute__((address_space(1))) v4i* as1_v4i_p;
typedef __attribute__((address_space(3))) v4i* as3_v4i_p;
__device__ inline void async16(const bf16_t* g, bf16_t* l) {
  __builtin_amdgcn_global_load_async_to_lds_b128(
      (as1_v4i_p)(unsigned long long)g,
      (as3_v4i_p)(unsigned)(unsigned long long)l,
      0, 0);
}
#endif

__device__ inline void wait_async_copies() {
#if HAVE_ASYNC_LDS
#if defined(__has_builtin) && __has_builtin(__builtin_amdgcn_s_wait_asynccnt)
  __builtin_amdgcn_s_wait_asynccnt(0);
#else
  asm volatile("s_wait_asynccnt 0x0" ::: "memory");
#endif
#endif
}

// -------------------------- generic bf16 WMMA GEMM -------------------------
// C[M,N] = A[M,K] @ B[K,N] (+bias). A: row-major bf16. B: kt-major swizzled:
//   addr(k,n) = ((kt*Ntiles + nt)*32 + lane)*16 + i
//   with kt=k/32, nt=n/16, lane = 16*((k%16)/8) + (n%16),
//        i = (k%32<16 ? 0 : 8) + (k%8)
// (matches the ISA 16-bit B-operand VGPR layout, so a lane's 16 values are
//  32 contiguous bytes). Output modes: 0=f32 rm, 1=bf16 rm, 2=B-swizzled bf16.

__device__ inline v16bf load_a16(const bf16_t* p0, const bf16_t* p1) {
  v8bf a = *(const v8bf*)p0;
  v8bf b = *(const v8bf*)p1;
  return __builtin_shufflevector(a, b, 0,1,2,3,4,5,6,7,8,9,10,11,12,13,14,15);
}

__device__ inline void stage_chunk(const bf16_t* __restrict__ Bsw, int Ntiles,
                                   int g, int kt0, bf16_t* dst, int tid) {
#pragma unroll
  for (int r = 0; r < CHUNK; ++r) {
    const bf16_t* src =
        Bsw + ((size_t)(kt0 + r) * Ntiles + (g << 3)) * 512 + tid * 16;
    bf16_t* d = dst + r * 4096 + tid * 16;
#if HAVE_ASYNC_LDS
    async16(src, d);
    async16(src + 8, d + 8);
#else
    *(float4*)d       = *(const float4*)src;
    *(float4*)(d + 8) = *(const float4*)(src + 8);
#endif
  }
}

__global__ void __launch_bounds__(256)
gemm_bf16_wmma(const bf16_t* __restrict__ A, int lda,
               const bf16_t* __restrict__ Bsw, int K, int Ntiles,
               const float* __restrict__ bias,
               int relu, int outMode,
               float* __restrict__ outF, bf16_t* __restrict__ outB, int ldc) {
  __shared__ __align__(32) bf16_t Bs[2][CHUNK * NT * 32 * 16];  // 2 x 16KB

  const int tid  = threadIdx.x;
  const int wave = tid >> 5;
  const int lane = tid & 31;
  const int ngroups = Ntiles >> 3;
  const int g      = blockIdx.x % ngroups;
  const int mpanel = blockIdx.x / ngroups;
  const int half = lane >> 4;
  const int nl   = lane & 15;

  const int row = mpanel * 128 + wave * 16 + nl;
  const bf16_t* arow = A + (size_t)row * lda + 8 * half;

  const int ktiles  = K >> 5;
  const int nchunks = ktiles / CHUNK;

  v8f acc[NT];
#pragma unroll
  for (int nt = 0; nt < NT; ++nt) {
    const float bv = bias ? bias[((g << 3) + nt) * 16 + nl] : 0.0f;
#pragma unroll
    for (int r = 0; r < 8; ++r) acc[nt][r] = bv;
  }

  stage_chunk(Bsw, Ntiles, g, 0, Bs[0], tid);
  wait_async_copies();
  __syncthreads();

  for (int c = 0; c < nchunks; ++c) {
    const bf16_t* buf = Bs[c & 1];
    if (c + 1 < nchunks)
      stage_chunk(Bsw, Ntiles, g, (c + 1) * CHUNK, Bs[(c + 1) & 1], tid);
#pragma unroll
    for (int ktl = 0; ktl < CHUNK; ++ktl) {
      const bf16_t* ap = arow + ((c * CHUNK + ktl) << 5);
      v16bf a = load_a16(ap, ap + 16);
      // Preload all 8 B tiles into distinct registers so the ds_loads batch
      // into a clause and pipeline against the WMMA stream.
      v16bf bvals[NT];
#pragma unroll
      for (int nt = 0; nt < NT; ++nt)
        bvals[nt] = *(const v16bf*)(buf + (ktl * NT + nt) * 512 + lane * 16);
#pragma unroll
      for (int nt = 0; nt < NT; ++nt)
        acc[nt] = __builtin_amdgcn_wmma_f32_16x16x32_bf16(
            false, a, false, bvals[nt], (short)0, acc[nt], false, false);
    }
    wait_async_copies();
    __syncthreads();
  }

#pragma unroll
  for (int nt = 0; nt < NT; ++nt) {
#pragma unroll
    for (int r = 0; r < 8; ++r) {
      const int m = mpanel * 128 + wave * 16 + r + (half << 3);
      const int n = (((g << 3) + nt) << 4) + nl;
      float v = acc[nt][r];
      if (relu) v = fmaxf(v, 0.0f);
      if (outMode == 0) {
        outF[(size_t)m * ldc + n] = v;
      } else if (outMode == 1) {
        outB[(size_t)m * ldc + n] = (bf16_t)v;
      } else {  // B-swizzled: row m becomes K of the consumer GEMM
        const int j   = (g << 3) + nt;
        const int ktp = m >> 5, kl = m & 31;
        const int hh  = (kl & 15) >> 3;
        const int ii  = (kl < 16 ? 0 : 8) + (kl & 7);
        const int lp  = (hh << 4) | nl;
        outB[(((size_t)(ktp * Ntiles + j) << 5) + lp) * 16 + ii] = (bf16_t)v;
      }
    }
  }
}

// ----------------------- weight -> B-swizzled bf16 (kt-major) --------------
__global__ void swizzle_weight(const float* __restrict__ W,
                               bf16_t* __restrict__ Bsw, int K, int N) {
  const int t = blockIdx.x * blockDim.x + threadIdx.x;
  const int Ntiles = N >> 4;
  const int total = Ntiles * (K >> 5) * 32;
  if (t >= total) return;
  const int lane = t & 31;
  const int idx  = t >> 5;
  const int nt   = idx % Ntiles;
  const int kt   = idx / Ntiles;
  const int halfv = lane >> 4, nl = lane & 15;
  bf16_t* dst = Bsw + (size_t)t * 16;
#pragma unroll
  for (int i = 0; i < 16; ++i) {
    const int k = (kt << 5) + (i < 8 ? 0 : 16) + (halfv << 3) + (i & 7);
    dst[i] = (bf16_t)W[(size_t)k * N + nt * 16 + nl];
  }
}

// ------------------- x: f32 -> bf16 copy + LDS transpose -------------------
__global__ void convert_transpose_x(const float* __restrict__ x,
                                    bf16_t* __restrict__ x_bf,
                                    bf16_t* __restrict__ xT_bf) {
  __shared__ float tile[32][33];
  const int tx = threadIdx.x, ty = threadIdx.y;
  const int c = blockIdx.x * 32 + tx;
  const int l = blockIdx.y * 32 + ty;
  const float v = x[(size_t)l * N_CLAUSES + c];
  x_bf[(size_t)l * N_CLAUSES + c] = (bf16_t)v;
  tile[ty][tx] = v;
  __syncthreads();
  const int cc = blockIdx.x * 32 + ty;
  const int ll = blockIdx.y * 32 + tx;
  xT_bf[(size_t)cc * N_LITS + ll] = (bf16_t)tile[tx][ty];
}

// ------------------------------ state init ---------------------------------
__global__ void init_state(const float* __restrict__ L_init,
                           const float* __restrict__ C_init,
                           bf16_t* __restrict__ Lh_bf, bf16_t* __restrict__ Ch_bf,
                           bf16_t* __restrict__ Cin,   bf16_t* __restrict__ Lin,
                           float* __restrict__ Lc,     float* __restrict__ Cc) {
  const int idx = blockIdx.x * blockDim.x + threadIdx.x;
  if (idx >= N_CLAUSES * DDIM) return;
  const int col = idx & 127, row = idx >> 7;
  const float rs = 0.08838834764831845f;  // 1/sqrt(128)
  const float ch = C_init[col] * rs;
  Ch_bf[idx] = (bf16_t)ch;
  Cin[(size_t)row * 256 + 128 + col] = (bf16_t)ch;
  Cc[idx] = 0.0f;
  if (row < N_LITS) {
    const float lh = L_init[col] * rs;
    Lh_bf[idx] = (bf16_t)lh;
    Lin[(size_t)row * 384 + 128 + col] = (bf16_t)lh;  // flip slot (rows equal)
    Lin[(size_t)row * 384 + 256 + col] = (bf16_t)lh;  // h slot
    Lc[idx] = 0.0f;
  }
}

// ------------------- LayerNorm LSTM cell (block = 1 row) -------------------
__device__ inline void blk_stats(float v, float* s1, float* s2, int t,
                                 float& mean, float& var) {
  s1[t] = v; s2[t] = v * v;
  __syncthreads();
  for (int off = 64; off; off >>= 1) {
    if (t < off) { s1[t] += s1[t + off]; s2[t] += s2[t + off]; }
    __syncthreads();
  }
  mean = s1[0] * (1.0f / 128.0f);
  var  = s2[0] * (1.0f / 128.0f) - mean * mean;
  __syncthreads();
}

__global__ void lstm_cell_kernel(const float* __restrict__ Z,   // [rows,512]
                                 float* __restrict__ Cst,       // [rows,128]
                                 const float* __restrict__ lng, const float* __restrict__ lnb,
                                 const float* __restrict__ cg,  const float* __restrict__ cb,
                                 bf16_t* __restrict__ Hbf,
                                 bf16_t* __restrict__ Cat, int catStride,
                                 bf16_t* __restrict__ Flip, int flipStride, int rowMask) {
  __shared__ float s1[128], s2[128];
  const int t = threadIdx.x, row = blockIdx.x;
  const float* zr = Z + (size_t)row * 512;
  float g4[4];
#pragma unroll
  for (int g = 0; g < 4; ++g) {
    const float v = zr[g * 128 + t];
    float m, var;
    blk_stats(v, s1, s2, t, m, var);
    g4[g] = (v - m) * rsqrtf(var + 1e-5f) * lng[g * 128 + t] + lnb[g * 128 + t];
  }
  const float f = 1.0f / (1.0f + expf(-g4[0]));
  const float i = 1.0f / (1.0f + expf(-g4[1]));
  const float o = 1.0f / (1.0f + expf(-g4[2]));
  const float u = tanhf(g4[3]);
  const float cpre = i * u + f * Cst[(size_t)row * 128 + t];
  float m, var;
  blk_stats(cpre, s1, s2, t, m, var);
  const float nc = (cpre - m) * rsqrtf(var + 1e-5f) * cg[t] + cb[t];
  Cst[(size_t)row * 128 + t] = nc;
  const float nh = o * tanhf(nc);
  Hbf[(size_t)row * 128 + t] = (bf16_t)nh;
  Cat[(size_t)row * catStride + t] = (bf16_t)nh;
  if (Flip) {
    const int fr = (row + N_VARS) & rowMask;
    Flip[(size_t)fr * flipStride + t] = (bf16_t)nh;
  }
}

// ----------------------------- final vote ----------------------------------
__global__ void vote_kernel(const float* __restrict__ H,       // [4096,128]
                            const float* __restrict__ vout,    // [128]
                            const float* __restrict__ voutb,   // [1]
                            const float* __restrict__ vbias,   // [1]
                            float* __restrict__ out) {
  __shared__ float s[256];
  const int g = blockIdx.x, t = threadIdx.x;
  const int r1 = g * 256 + t;
  const int r2 = N_VARS + g * 256 + t;
  float d1 = 0.0f, d2 = 0.0f;
  for (int j = 0; j < 128; ++j) {
    const float w = vout[j];
    d1 += fmaxf(H[(size_t)r1 * 128 + j], 0.0f) * w;
    d2 += fmaxf(H[(size_t)r2 * 128 + j], 0.0f) * w;
  }
  s[t] = d1 + d2 + 2.0f * voutb[0];
  __syncthreads();
  for (int off = 128; off; off >>= 1) {
    if (t < off) s[t] += s[t + off];
    __syncthreads();
  }
  if (t == 0) out[g] = s[0] * (1.0f / 512.0f) + vbias[0];
}

// ---------------------------------------------------------------------------
extern "C" void kernel_launch(void* const* d_in, const int* in_sizes, int n_in,
                              void* d_out, int out_size, void* d_ws, size_t ws_size,
                              hipStream_t stream) {
  const float* x        = (const float*)d_in[0];
  const float* L_init   = (const float*)d_in[1];
  const float* C_init   = (const float*)d_in[2];
  const float* LC_W     = (const float*)d_in[3];
  const float* LC_b     = (const float*)d_in[4];
  const float* CL_W     = (const float*)d_in[5];
  const float* CL_b     = (const float*)d_in[6];
  const float* Lu_W     = (const float*)d_in[7];
  const float* Lu_ln_g  = (const float*)d_in[8];
  const float* Lu_ln_b  = (const float*)d_in[9];
  const float* Lu_cln_g = (const float*)d_in[10];
  const float* Lu_cln_b = (const float*)d_in[11];
  const float* Cu_W     = (const float*)d_in[12];
  const float* Cu_ln_g  = (const float*)d_in[13];
  const float* Cu_ln_b  = (const float*)d_in[14];
  const float* Cu_cln_g = (const float*)d_in[15];
  const float* Cu_cln_b = (const float*)d_in[16];
  const float* vote_W   = (const float*)d_in[17];
  const float* vote_b   = (const float*)d_in[18];
  const float* vote_oW  = (const float*)d_in[19];
  const float* vote_ob  = (const float*)d_in[20];
  const float* vote_bias= (const float*)d_in[21];
  float* out = (float*)d_out;

  // ---- workspace carve-up ----
  char* p = (char*)d_ws;
  auto alloc = [&](size_t bytes) -> char* {
    char* r = p; p += (bytes + 255) & ~(size_t)255; return r;
  };
  bf16_t* x_bf  = (bf16_t*)alloc((size_t)N_LITS * N_CLAUSES * 2);
  bf16_t* xT_bf = (bf16_t*)alloc((size_t)N_LITS * N_CLAUSES * 2);
  bf16_t* Cin   = (bf16_t*)alloc((size_t)N_CLAUSES * 256 * 2);
  bf16_t* Lin   = (bf16_t*)alloc((size_t)N_LITS * 384 * 2);
  float*  Zc    = (float*) alloc((size_t)N_CLAUSES * 512 * 4);
  float*  Zl    = (float*) alloc((size_t)N_LITS * 512 * 4);
  float*  Cc    = (float*) alloc((size_t)N_CLAUSES * 128 * 4);
  float*  Lc    = (float*) alloc((size_t)N_LITS * 128 * 4);
  bf16_t* Ch_bf = (bf16_t*)alloc((size_t)N_CLAUSES * 128 * 2);
  bf16_t* Lh_bf = (bf16_t*)alloc((size_t)N_LITS * 128 * 2);
  bf16_t* T1    = (bf16_t*)alloc((size_t)N_LITS * 128 * 2);
  bf16_t* T2    = (bf16_t*)alloc((size_t)N_LITS * 128 * 2);
  bf16_t* U1    = (bf16_t*)alloc((size_t)N_CLAUSES * 128 * 2);
  bf16_t* U2    = (bf16_t*)alloc((size_t)N_CLAUSES * 128 * 2);
  bf16_t* MLsw  = (bf16_t*)alloc((size_t)N_LITS * 128 * 2);
  bf16_t* MCsw  = (bf16_t*)alloc((size_t)N_CLAUSES * 128 * 2);
  float*  V2    = (float*) alloc((size_t)N_LITS * 128 * 4);
  bf16_t* LCWs[3], *CLWs[3], *VWs[2];
  for (int i = 0; i < 3; ++i) LCWs[i] = (bf16_t*)alloc(128 * 128 * 2);
  for (int i = 0; i < 3; ++i) CLWs[i] = (bf16_t*)alloc(128 * 128 * 2);
  for (int i = 0; i < 2; ++i) VWs[i]  = (bf16_t*)alloc(128 * 128 * 2);
  bf16_t* CuWsw = (bf16_t*)alloc(256 * 512 * 2);
  bf16_t* LuWsw = (bf16_t*)alloc(384 * 512 * 2);

  // ---- prep: swizzle weights, convert/transpose x, init state ----
  auto swz = [&](const float* W, bf16_t* dst, int K, int N) {
    int total = (N >> 4) * (K >> 5) * 32;
    swizzle_weight<<<(total + 255) / 256, 256, 0, stream>>>(W, dst, K, N);
  };
  for (int i = 0; i < 3; ++i) swz(LC_W + (size_t)i * 128 * 128, LCWs[i], 128, 128);
  for (int i = 0; i < 3; ++i) swz(CL_W + (size_t)i * 128 * 128, CLWs[i], 128, 128);
  for (int i = 0; i < 2; ++i) swz(vote_W + (size_t)i * 128 * 128, VWs[i], 128, 128);
  swz(Cu_W, CuWsw, 256, 512);
  swz(Lu_W, LuWsw, 384, 512);

  convert_transpose_x<<<dim3(N_CLAUSES / 32, N_LITS / 32), dim3(32, 32), 0, stream>>>(
      x, x_bf, xT_bf);
  init_state<<<(N_CLAUSES * 128 + 255) / 256, 256, 0, stream>>>(
      L_init, C_init, Lh_bf, Ch_bf, Cin, Lin, Lc, Cc);

  // C[M,N] = A @ B(+bias); grid = (M/128) * (N/128)
  auto gemm = [&](const bf16_t* A, int lda, const bf16_t* Bsw, int K,
                  const float* bias, int M, int N, int relu, int mode,
                  float* oF, bf16_t* oB, int ldc) {
    int blocks = (M >> 7) * (N >> 7);
    gemm_bf16_wmma<<<blocks, 256, 0, stream>>>(
        A, lda, Bsw, K, N >> 4, bias, relu, mode, oF, oB, ldc);
  };

  // ---- message-passing rounds ----
  for (int r = 0; r < N_ROUNDS; ++r) {
    // MLP(Lh): 128->128->128->128, relu between, last layer emitted B-swizzled
    gemm(Lh_bf, 128, LCWs[0], 128, LC_b + 0,   N_LITS, 128, 1, 1, nullptr, T1, 128);
    gemm(T1,    128, LCWs[1], 128, LC_b + 128, N_LITS, 128, 1, 1, nullptr, T2, 128);
    gemm(T2,    128, LCWs[2], 128, LC_b + 256, N_LITS, 128, 0, 2, nullptr, MLsw, 0);
    // LC_msgs = x.T @ ML -> Cin[:, :128]
    gemm(xT_bf, N_LITS, MLsw, N_LITS, nullptr, N_CLAUSES, 128, 0, 1, nullptr, Cin, 256);
    // clause LSTM linear: Zc = [LC_msgs | Ch] @ Cu_W (bias-free)
    gemm(Cin, 256, CuWsw, 256, nullptr, N_CLAUSES, 512, 0, 0, Zc, nullptr, 512);
    lstm_cell_kernel<<<N_CLAUSES, 128, 0, stream>>>(
        Zc, Cc, Cu_ln_g, Cu_ln_b, Cu_cln_g, Cu_cln_b,
        Ch_bf, Cin + 128, 256, nullptr, 0, 0);
    // MLP(Ch2)
    gemm(Ch_bf, 128, CLWs[0], 128, CL_b + 0,   N_CLAUSES, 128, 1, 1, nullptr, U1, 128);
    gemm(U1,    128, CLWs[1], 128, CL_b + 128, N_CLAUSES, 128, 1, 1, nullptr, U2, 128);
    gemm(U2,    128, CLWs[2], 128, CL_b + 256, N_CLAUSES, 128, 0, 2, nullptr, MCsw, 0);
    // CL_msgs = x @ MC -> Lin[:, :128]
    gemm(x_bf, N_CLAUSES, MCsw, N_CLAUSES, nullptr, N_LITS, 128, 0, 1, nullptr, Lin, 384);
    // literal LSTM linear: Zl = [CL_msgs | flipped | Lh] @ Lu_W
    gemm(Lin, 384, LuWsw, 384, nullptr, N_LITS, 512, 0, 0, Zl, nullptr, 512);
    lstm_cell_kernel<<<N_LITS, 128, 0, stream>>>(
        Zl, Lc, Lu_ln_g, Lu_ln_b, Lu_cln_g, Lu_cln_b,
        Lh_bf, Lin + 256, 384, Lin + 128, 384, N_LITS - 1);
  }

  // ---- vote head ----
  gemm(Lh_bf, 128, VWs[0], 128, vote_b + 0,   N_LITS, 128, 1, 1, nullptr, T1, 128);
  gemm(T1,    128, VWs[1], 128, vote_b + 128, N_LITS, 128, 0, 0, V2, nullptr, 128);
  vote_kernel<<<8, 256, 0, stream>>>(V2, vote_oW, vote_ob, vote_bias, out);

  (void)in_sizes; (void)n_in; (void)out_size; (void)ws_size;
}
